// FilterGNN_57904749084722
// MI455X (gfx1250) — matrix operand order repaired and checked by reading, not compiled, not run on hardware
//
#include <hip/hip_runtime.h>
#include <hip/hip_bf16.h>

typedef __attribute__((ext_vector_type(16))) __bf16 v16bf;
typedef __attribute__((ext_vector_type(8)))  __bf16 v8bf;
typedef __attribute__((ext_vector_type(8)))  float  v8f;

#define H1_PITCH 520   // 512 + 8 pad: row stride = 260 words -> 4-bank rotation, conflict-free b128 reads
#define H2_PITCH 264   // 256 + 8 pad

__device__ __forceinline__ __bf16 f2bf(float f) {
    return (__bf16)f;   // hardware v_cvt (RNE) on gfx1250
}

// ---------------------------------------------------------------------------
// Pre-pass: convert W1/W2/W3 (f32) to bf16 in workspace
// ---------------------------------------------------------------------------
__global__ void fgnn_wconv(const float* __restrict__ W1, const float* __restrict__ W2,
                           const float* __restrict__ W3,
                           __bf16* __restrict__ w1, __bf16* __restrict__ w2,
                           __bf16* __restrict__ w3) {
    int i = blockIdx.x * blockDim.x + threadIdx.x;
    if (i < 512 * 256) w1[i] = f2bf(W1[i]);
    if (i < 256 * 512) w2[i] = f2bf(W2[i]);
    if (i < 128 * 256) w3[i] = f2bf(W3[i]);
}

// ---------------------------------------------------------------------------
// Zero-init output accumulator and per-node edge counts
// ---------------------------------------------------------------------------
__global__ void fgnn_zero(float* __restrict__ out, int* __restrict__ cnt,
                          int nOut, int nCnt) {
    int i = blockIdx.x * blockDim.x + threadIdx.x;
    if (i < nOut) out[i] = 0.0f;
    if (i < nCnt) cnt[i] = 0;
}

// ---------------------------------------------------------------------------
// Fused 3-layer MLP with WMMA bf16. One block (8 wave32) per 16-row tile.
//   stage1: h1 = tanh(x @ W1^T + b1)   [16,512]  (waves split N: 4 tiles each)
//   stage2: h2 = tanh(h1 @ W2^T + b2)  [16,256]  (2 tiles each)
//   stage3: h  = h2 @ W3^T + b3        [16,128]  (1 tile each) -> global
// ---------------------------------------------------------------------------
__global__ __launch_bounds__(256) void fgnn_mlp_wmma(
    const float* __restrict__ x,
    const __bf16* __restrict__ w1,
    const __bf16* __restrict__ w2,
    const __bf16* __restrict__ w3,
    const float* __restrict__ b1,
    const float* __restrict__ b2,
    const float* __restrict__ b3,
    float* __restrict__ h_out, int Nn)
{
    __shared__ __align__(16) __bf16 lds_h1[16 * H1_PITCH];
    __shared__ __align__(16) __bf16 lds_h2[16 * H2_PITCH];

    const int wave = threadIdx.x >> 5;
    const int lane = threadIdx.x & 31;
    const int m    = lane & 15;       // M (A/C) or N (B) index of this lane
    const int half = lane >> 4;       // half-wave selector in K
    const int row0 = blockIdx.x * 16;
    const int row  = row0 + m;
    const int rowc = row < Nn ? row : (Nn - 1);

    // warm GL2 with the weight working set (global_prefetch_b8)
    if (lane == 0) {
        __builtin_prefetch(w2 + wave * 8192, 0, 3);
        __builtin_prefetch(w3 + wave * 4096, 0, 3);
    }

    // ---------------- Stage 1 ----------------
    // Build 8 bf16 A-fragments (16x32 each, K=256) straight from f32 x.
    // A layout: lane m = M; elements 0..15 -> K = k0+16*half+e (K-contiguous)
    v16bf af[8];
#pragma unroll
    for (int c = 0; c < 8; ++c) {
        const int k0 = c * 32;
        const float4* plo = (const float4*)(x + (size_t)rowc * 256 + k0 + 16 * half);
        const float4* phi = (const float4*)(x + (size_t)rowc * 256 + k0 + 16 * half + 8);
        float4 l0 = plo[0], l1 = plo[1];
        float4 h0 = phi[0], h1 = phi[1];
        v16bf a;
        a[0]  = f2bf(l0.x); a[1]  = f2bf(l0.y); a[2]  = f2bf(l0.z); a[3]  = f2bf(l0.w);
        a[4]  = f2bf(l1.x); a[5]  = f2bf(l1.y); a[6]  = f2bf(l1.z); a[7]  = f2bf(l1.w);
        a[8]  = f2bf(h0.x); a[9]  = f2bf(h0.y); a[10] = f2bf(h0.z); a[11] = f2bf(h0.w);
        a[12] = f2bf(h1.x); a[13] = f2bf(h1.y); a[14] = f2bf(h1.z); a[15] = f2bf(h1.w);
        af[c] = a;
    }
#pragma unroll
    for (int j = 0; j < 4; ++j) {
        const int n0 = (wave * 4 + j) * 16;
        v8f acc = {};
#pragma unroll
        for (int c = 0; c < 8; ++c) {
            const int k0 = c * 32;
            // B layout: lane m = N; elements 0..15 -> K = k0 + 16*half + e (K-contiguous)
            v16bf bu = *(const v16bf*)(w1 + (size_t)(n0 + m) * 256 + k0 + 16 * half);
            acc = __builtin_amdgcn_wmma_f32_16x16x32_bf16(
                false, af[c], false, bu, (short)0, acc, false, false);
        }
        const float bias = b1[n0 + m];
#pragma unroll
        for (int r = 0; r < 8; ++r) {   // C/D layout: vgpr r -> M = r + 8*half, N = m
            lds_h1[(r + 8 * half) * H1_PITCH + n0 + m] = f2bf(tanhf(acc[r] + bias));
        }
    }
    __syncthreads();

    // ---------------- Stage 2 ----------------
    {
        const int na = (wave * 2 + 0) * 16;
        const int nb = (wave * 2 + 1) * 16;
        v8f acc2a = {}, acc2b = {};
#pragma unroll
        for (int c = 0; c < 16; ++c) {
            const int k0 = c * 32;
            v8bf lo = *(const v8bf*)(&lds_h1[m * H1_PITCH + k0 + 16 * half]);
            v8bf hi = *(const v8bf*)(&lds_h1[m * H1_PITCH + k0 + 16 * half + 8]);
            v16bf a = __builtin_shufflevector(lo, hi, 0,1,2,3,4,5,6,7,8,9,10,11,12,13,14,15);
            v16bf ba = *(const v16bf*)(w2 + (size_t)(na + m) * 512 + k0 + 16 * half);
            v16bf bb = *(const v16bf*)(w2 + (size_t)(nb + m) * 512 + k0 + 16 * half);
            acc2a = __builtin_amdgcn_wmma_f32_16x16x32_bf16(
                false, a, false, ba, (short)0, acc2a, false, false);
            acc2b = __builtin_amdgcn_wmma_f32_16x16x32_bf16(
                false, a, false, bb, (short)0, acc2b, false, false);
        }
        const float biasA = b2[na + m];
        const float biasB = b2[nb + m];
#pragma unroll
        for (int r = 0; r < 8; ++r) {
            lds_h2[(r + 8 * half) * H2_PITCH + na + m] = f2bf(tanhf(acc2a[r] + biasA));
            lds_h2[(r + 8 * half) * H2_PITCH + nb + m] = f2bf(tanhf(acc2b[r] + biasB));
        }
    }
    __syncthreads();

    // ---------------- Stage 3 ----------------
    {
        const int n0 = wave * 16;
        v8f acc3 = {};
#pragma unroll
        for (int c = 0; c < 8; ++c) {
            const int k0 = c * 32;
            v8bf lo = *(const v8bf*)(&lds_h2[m * H2_PITCH + k0 + 16 * half]);
            v8bf hi = *(const v8bf*)(&lds_h2[m * H2_PITCH + k0 + 16 * half + 8]);
            v16bf a = __builtin_shufflevector(lo, hi, 0,1,2,3,4,5,6,7,8,9,10,11,12,13,14,15);
            v16bf bu = *(const v16bf*)(w3 + (size_t)(n0 + m) * 256 + k0 + 16 * half);
            acc3 = __builtin_amdgcn_wmma_f32_16x16x32_bf16(
                false, a, false, bu, (short)0, acc3, false, false);
        }
        const float bias = b3[n0 + m];
#pragma unroll
        for (int r = 0; r < 8; ++r) {
            const int rr = row0 + r + 8 * half;
            if (rr < Nn) h_out[(size_t)rr * 128 + n0 + m] = acc3[r] + bias;
        }
    }
}

// ---------------------------------------------------------------------------
// Per-edge gather + gate + atomic scatter-add. 32 threads per edge, 4 feats each.
// ---------------------------------------------------------------------------
__global__ __launch_bounds__(256) void fgnn_scatter(
    const float* __restrict__ h, const float* __restrict__ alpha,
    const int* __restrict__ ei, float* __restrict__ out,
    int* __restrict__ cnt, int E)
{
    long long tid = (long long)blockIdx.x * blockDim.x + threadIdx.x;
    int e = (int)(tid >> 5);
    if (e >= E) return;
    int g   = (int)(tid & 31);
    int src = ei[e];
    int dst = ei[E + e];
    if (g == 0) atomicAdd(&cnt[dst], 1);          // mean counts ALL edges
    float a    = alpha[e];
    float gate = 2.0f * fmaxf(a - 0.5f, 0.0f);
    if (gate > 0.0f) {
        const float4 hv = *(const float4*)(h + (size_t)src * 128 + g * 4);
        float* o = out + (size_t)dst * 128 + g * 4;
        unsafeAtomicAdd(o + 0, hv.x * gate);      // -> global_atomic_add_f32
        unsafeAtomicAdd(o + 1, hv.y * gate);
        unsafeAtomicAdd(o + 2, hv.z * gate);
        unsafeAtomicAdd(o + 3, hv.w * gate);
    }
}

// ---------------------------------------------------------------------------
// Divide by per-node edge count (0-edge nodes stay 0)
// ---------------------------------------------------------------------------
__global__ void fgnn_final(float* __restrict__ out, const int* __restrict__ cnt, int Nn) {
    int i = blockIdx.x * blockDim.x + threadIdx.x;
    if (i >= Nn * 128) return;
    int n = i >> 7;
    float c = (float)cnt[n];
    out[i] *= (1.0f / fmaxf(c, 1.0f));
}

extern "C" void kernel_launch(void* const* d_in, const int* in_sizes, int n_in,
                              void* d_out, int out_size, void* d_ws, size_t ws_size,
                              hipStream_t stream) {
    const float* x     = (const float*)d_in[0];
    const float* alpha = (const float*)d_in[1];
    const int*   ei    = (const int*)d_in[2];
    // d_in[3] = norm (unused by the reference math)
    const float* W1 = (const float*)d_in[4];
    const float* b1 = (const float*)d_in[5];
    const float* W2 = (const float*)d_in[6];
    const float* b2 = (const float*)d_in[7];
    const float* W3 = (const float*)d_in[8];
    const float* b3 = (const float*)d_in[9];

    const int Nn = in_sizes[0] / 256;   // IN = 256
    const int E  = in_sizes[1];         // alpha has E elements

    // workspace layout (256B aligned): bf16 weights | int counts | f32 h
    char* ws = (char*)d_ws;
    __bf16* w1bf = (__bf16*)(ws);
    __bf16* w2bf = (__bf16*)(ws + 262144);
    __bf16* w3bf = (__bf16*)(ws + 524288);
    int*   cnt = (int*)(ws + 589824);
    size_t hoff = 589824 + (((size_t)Nn * 4 + 255) / 256) * 256;
    float* h   = (float*)(ws + hoff);
    float* out = (float*)d_out;

    fgnn_wconv<<<(131072 + 255) / 256, 256, 0, stream>>>(W1, W2, W3, w1bf, w2bf, w3bf);

    const int nOut = Nn * 128;
    fgnn_zero<<<(nOut + 255) / 256, 256, 0, stream>>>(out, cnt, nOut, Nn);

    fgnn_mlp_wmma<<<(Nn + 15) / 16, 256, 0, stream>>>(x, w1bf, w2bf, w3bf, b1, b2, b3, h, Nn);

    long long st = (long long)E * 32;
    fgnn_scatter<<<(unsigned)((st + 255) / 256), 256, 0, stream>>>(h, alpha, ei, out, cnt, E);

    fgnn_final<<<(nOut + 255) / 256, 256, 0, stream>>>(out, cnt, Nn);
}
